// Seq2Seq_72928544686496
// MI455X (gfx1250) — compile-verified, matrix operand order checked by reading, not run on hardware
//
#include <hip/hip_runtime.h>

// ---------------------------------------------------------------------------
// Seq2Seq (LSTM encoder + MHA-attention LSTM decoder) for MI455X / gfx1250.
// All matmuls use v_wmma_f32_16x16x32_bf16 (B=16 == WMMA M tile).
// Encoder scan double-buffers its precomputed-gate slabs into LDS with
// GLOBAL_LOAD_ASYNC_TO_LDS_B128 (ASYNCcnt) when the toolchain exposes it.
// ---------------------------------------------------------------------------

constexpr int kB  = 16;
constexpr int kS  = 1024;
constexpr int kT  = 128;
constexpr int kV  = 10000;
constexpr int kE  = 256;
constexpr int kH  = 128;
constexpr int kNH = 8;
constexpr int kDH = 16;
constexpr int kG  = 512;   // 4*H

typedef __bf16 bf16;
typedef __attribute__((ext_vector_type(8)))  bf16  bf16x8;
typedef __attribute__((ext_vector_type(16))) bf16  bf16x16;
typedef __attribute__((ext_vector_type(8)))  float f32x8;
typedef __attribute__((ext_vector_type(4)))  int   i32x4;

#if defined(__AMDGCN__) && \
    __has_builtin(__builtin_amdgcn_global_load_async_to_lds_b128) && \
    __has_builtin(__builtin_amdgcn_s_wait_asynccnt)
#define USE_ASYNC_LDS 1
// Builtin parameter types: pointer-to-int4 in global (AS1) / LDS (AS3).
typedef __attribute__((address_space(1))) i32x4* gas_i32x4p;
typedef __attribute__((address_space(3))) i32x4* las_i32x4p;
#else
#define USE_ASYNC_LDS 0
#endif

// Dynamic-LDS layout for the encoder scan (same constant on host & device):
// gxbuf: 2*16*512 f32 | gates: 16*512 f32 | cbuf: 16*128 f32 | h_bf: 16*128 bf16
constexpr unsigned kSmemEnc =
    (2 * kB * kG + kB * kG + kB * kH) * 4 + kB * kH * 2;  // 110592 B

// ---------------- device helpers ----------------

__device__ __forceinline__ bf16 f2bf(float f) { return (bf16)f; }
__device__ __forceinline__ float bf2f(bf16 b) { return (float)b; }
__device__ __forceinline__ float sigm(float x) { return 1.0f / (1.0f + __expf(-x)); }

__device__ __forceinline__ f32x8 wmma_bf(bf16x16 a, bf16x16 b, f32x8 c) {
  return __builtin_amdgcn_wmma_f32_16x16x32_bf16(false, a, false, b, (short)0, c,
                                                 false, false);
}

// A/B fragment loader. Row-major source (A: M x K activations, B: W stored as
// (N,K) so columns of B == rows of W). ISA 16-bit layout: lane L<16 holds row
// (L&15) with K in {k0..k0+7, k0+16..k0+23}; lane L>=16 holds K shifted by 8.
// rowbase must already point at element [row][0] for this lane's row.
__device__ __forceinline__ bf16x16 frag_ld(const bf16* rowbase, int k0) {
  const int hi = (threadIdx.x >> 4) & 1;
  const bf16* p = rowbase + k0 + hi * 8;
  bf16x8 lo = *(const bf16x8*)(p);
  bf16x8 hh = *(const bf16x8*)(p + 16);
  bf16x16 r;
#pragma unroll
  for (int i = 0; i < 8; i++) { r[i] = lo[i]; r[8 + i] = hh[i]; }
  return r;
}

#if USE_ASYNC_LDS
// Copy 32B per thread (2 x b128) from global to LDS asynchronously.
__device__ __forceinline__ void async_slab(const float* gsrc, float* ldst, int tid) {
  const float* g = gsrc + tid * 8;
  float* l = ldst + tid * 8;
  __builtin_amdgcn_global_load_async_to_lds_b128((gas_i32x4p)g, (las_i32x4p)l, 0, 0);
  __builtin_amdgcn_global_load_async_to_lds_b128((gas_i32x4p)(g + 4),
                                                 (las_i32x4p)(l + 4), 0, 0);
}
#endif

// ---------------- small utility kernels ----------------

__global__ void k_cvt(bf16* __restrict__ d, const float* __restrict__ s, int n) {
  int i = blockIdx.x * blockDim.x + threadIdx.x;
  if (i < n) d[i] = f2bf(s[i]);
}

__global__ void k_addv(float* __restrict__ d, const float* __restrict__ a,
                       const float* __restrict__ b, int n) {
  int i = blockIdx.x * blockDim.x + threadIdx.x;
  if (i < n) d[i] = a[i] + b[i];
}

__global__ void k_zero_out0(float* __restrict__ out) {
  int i = blockIdx.x * blockDim.x + threadIdx.x;
  if (i < kB * kV) {
    int b = i / kV, v = i % kV;
    out[(size_t)b * kT * kV + v] = 0.0f;
  }
}

__global__ void k_init_tok(int* __restrict__ cur, const int* __restrict__ trg) {
  if (threadIdx.x < kB) cur[threadIdx.x] = trg[threadIdx.x * kT];
}

// ---------------- encoder: precompute x @ Wih^T for all timesteps ----------

// grid = S blocks, 256 threads (8 waves, 4 N-tiles each). gx layout (S,B,4H).
__global__ void __launch_bounds__(256) k_embgx(
    const int* __restrict__ src, const float* __restrict__ emb,
    const bf16* __restrict__ Wih, const float* __restrict__ bias,
    float* __restrict__ gx) {
  __shared__ bf16 xe[kB][kE];  // gathered+converted embedding rows (8 KB)
  const int s = blockIdx.x;
  const int tid = threadIdx.x;
  for (int i = tid; i < kB * kE; i += 256) {
    const int b = i >> 8, col = i & 255;
    xe[b][col] = f2bf(emb[(size_t)src[b * kS + s] * kE + col]);
  }
  __syncthreads();

  const int wave = tid >> 5;
  const int lane = tid & 31;
  const int rowm = lane & 15, n = lane & 15, m0 = (lane >> 4) * 8;
  const bf16* arow = &xe[rowm][0];

  f32x8 acc[4] = {{}, {}, {}, {}};
#pragma unroll
  for (int kk = 0; kk < kE; kk += 32) {
    bf16x16 a = frag_ld(arow, kk);
#pragma unroll
    for (int t = 0; t < 4; t++) {
      const int nt = wave * 4 + t;
      const bf16* brow = Wih + (size_t)(nt * 16 + rowm) * kE;
      acc[t] = wmma_bf(a, frag_ld(brow, kk), acc[t]);
    }
  }
  float* outp = gx + (size_t)s * kB * kG;
#pragma unroll
  for (int t = 0; t < 4; t++) {
    const int col = (wave * 4 + t) * 16 + n;
    const float bv = bias[col];
#pragma unroll
    for (int r = 0; r < 8; r++) outp[(size_t)(m0 + r) * kG + col] = acc[t][r] + bv;
  }
}

// ---------------- encoder: serial scan (1 workgroup, 32 waves) -------------

__global__ void __launch_bounds__(1024) k_enc_scan(
    const float* __restrict__ gx, const bf16* __restrict__ Whh,
    bf16* __restrict__ enc_out, bf16* __restrict__ h_out, float* __restrict__ c_out) {
  extern __shared__ char smem[];
  float* gxbuf = (float*)smem;               // 2 * 16*512 f32 (async ping-pong)
  float* gates = gxbuf + 2 * kB * kG;        // 16*512 f32
  float* cbuf  = gates + kB * kG;            // 16*128 f32
  bf16*  h_bf  = (bf16*)(cbuf + kB * kH);    // 16*128 bf16

  const int tid = threadIdx.x;
  for (int i = tid; i < kB * kH; i += 1024) {
    h_bf[i] = f2bf(0.0f);
    cbuf[i] = 0.0f;
  }

  const int wave = tid >> 5;  // one N-tile of 4H per wave
  const int lane = tid & 31;
  const int rowm = lane & 15, n = lane & 15, m0 = (lane >> 4) * 8;
  const bf16* hrow = h_bf + rowm * kH;

  // Whh B-fragments are invariant across all 1024 steps: hoist into VGPRs.
  bf16x16 bfrag[4];
  {
    const bf16* brow = Whh + (size_t)(wave * 16 + rowm) * kH;
#pragma unroll
    for (int k4 = 0; k4 < 4; k4++) bfrag[k4] = frag_ld(brow, k4 * 32);
  }

#if USE_ASYNC_LDS
  async_slab(gx, gxbuf, tid);  // slab 0 -> buffer 0
#endif
  __syncthreads();

  for (int s = 0; s < kS; s++) {
#if USE_ASYNC_LDS
    __builtin_amdgcn_s_wait_asynccnt(0);
    __syncthreads();
    const float* gxs = gxbuf + (s & 1) * kB * kG;
    if (s + 1 < kS)
      async_slab(gx + (size_t)(s + 1) * kB * kG, gxbuf + ((s + 1) & 1) * kB * kG, tid);
#else
    const float* gxs = gx + (size_t)s * kB * kG;
    if (s + 1 < kS)
      __builtin_prefetch(gx + (size_t)(s + 1) * kB * kG + tid * 8, 0, 1);
#endif
    f32x8 acc;
#pragma unroll
    for (int r = 0; r < 8; r++) acc[r] = gxs[(size_t)(m0 + r) * kG + wave * 16 + n];
#pragma unroll
    for (int k4 = 0; k4 < 4; k4++)
      acc = wmma_bf(frag_ld(hrow, k4 * 32), bfrag[k4], acc);

#pragma unroll
    for (int r = 0; r < 8; r++) gates[(m0 + r) * kG + wave * 16 + n] = acc[r];
    __syncthreads();

    for (int e = tid; e < kB * kH; e += 1024) {
      const int b = e >> 7, j = e & 127;
      const float ig = gates[b * kG + j], fg = gates[b * kG + kH + j];
      const float gg = gates[b * kG + 2 * kH + j], og = gates[b * kG + 3 * kH + j];
      float c = cbuf[e];
      c = sigm(fg) * c + sigm(ig) * tanhf(gg);
      const float h = sigm(og) * tanhf(c);
      cbuf[e] = c;
      const bf16 hb = f2bf(h);
      h_bf[e] = hb;
      enc_out[((size_t)s * kB + b) * kH + j] = hb;
    }
    __syncthreads();
  }
  for (int e = tid; e < kB * kH; e += 1024) {
    h_out[e] = h_bf[e];
    c_out[e] = cbuf[e];
  }
}

// ---------------- K/V projections (parallel over S) ------------------------

__global__ void __launch_bounds__(256) k_kv(
    const bf16* __restrict__ enc_out, const bf16* __restrict__ Wk,
    const bf16* __restrict__ Wv, const float* __restrict__ bk,
    const float* __restrict__ bv, bf16* __restrict__ kh, bf16* __restrict__ vh) {
  const int s = blockIdx.x;
  const int wave = threadIdx.x >> 5;   // 8 waves == 8 N-tiles of H
  const int lane = threadIdx.x & 31;
  const int rowm = lane & 15, n = lane & 15, m0 = (lane >> 4) * 8;
  const bf16* arow = enc_out + ((size_t)s * kB + rowm) * kH;
  const bf16* bkrow = Wk + (size_t)(wave * 16 + rowm) * kH;
  const bf16* bvrow = Wv + (size_t)(wave * 16 + rowm) * kH;

  f32x8 ak = {}, av = {};
#pragma unroll
  for (int kk = 0; kk < kH; kk += 32) {
    bf16x16 a = frag_ld(arow, kk);
    ak = wmma_bf(a, frag_ld(bkrow, kk), ak);
    av = wmma_bf(a, frag_ld(bvrow, kk), av);
  }
  const int col = wave * 16 + n;
  const float bkc = bk[col], bvc = bv[col];
#pragma unroll
  for (int r = 0; r < 8; r++) {
    kh[((size_t)s * kB + m0 + r) * kH + col] = f2bf(ak[r] + bkc);
    vh[((size_t)s * kB + m0 + r) * kH + col] = f2bf(av[r] + bvc);
  }
}

// ---------------- decoder: q projection ------------------------------------

__global__ void __launch_bounds__(256) k_q(
    const bf16* __restrict__ hbf, const bf16* __restrict__ Wq,
    const float* __restrict__ bq, float* __restrict__ q) {
  const int wave = threadIdx.x >> 5;
  const int lane = threadIdx.x & 31;
  const int rowm = lane & 15, n = lane & 15, m0 = (lane >> 4) * 8;
  const bf16* arow = hbf + rowm * kH;
  const bf16* brow = Wq + (size_t)(wave * 16 + rowm) * kH;
  f32x8 acc = {};
#pragma unroll
  for (int kk = 0; kk < kH; kk += 32)
    acc = wmma_bf(frag_ld(arow, kk), frag_ld(brow, kk), acc);
  const int col = wave * 16 + n;
  const float bb = bq[col];
#pragma unroll
  for (int r = 0; r < 8; r++) q[(m0 + r) * kH + col] = acc[r] + bb;
}

// ---------------- decoder: per-(b,head) attention --------------------------

__global__ void __launch_bounds__(256) k_attn(
    const float* __restrict__ q, const bf16* __restrict__ kh,
    const bf16* __restrict__ vh, float* __restrict__ ctx) {
  __shared__ float sc[kS];
  __shared__ float red[256];
  __shared__ float part[kDH][17];
  const int b = blockIdx.x >> 3, hd = blockIdx.x & 7;
  const int tid = threadIdx.x;

  float qv[kDH];
#pragma unroll
  for (int d = 0; d < kDH; d++) qv[d] = q[b * kH + hd * kDH + d];
  const float scale = 0.25f;  // 1/sqrt(16)

  for (int s0 = tid; s0 < kS; s0 += 256) {
    const bf16* kp = kh + ((size_t)s0 * kB + b) * kH + hd * kDH;
    float a = 0.0f;
#pragma unroll
    for (int d = 0; d < kDH; d++) a += qv[d] * bf2f(kp[d]);
    sc[s0] = a * scale;
  }
  __syncthreads();

  float m = -1e30f;
  for (int s0 = tid; s0 < kS; s0 += 256) m = fmaxf(m, sc[s0]);
  red[tid] = m;
  __syncthreads();
  for (int o = 128; o > 0; o >>= 1) {
    if (tid < o) red[tid] = fmaxf(red[tid], red[tid + o]);
    __syncthreads();
  }
  m = red[0];
  __syncthreads();

  float ssum = 0.0f;
  for (int s0 = tid; s0 < kS; s0 += 256) {
    const float e = __expf(sc[s0] - m);
    sc[s0] = e;
    ssum += e;
  }
  red[tid] = ssum;
  __syncthreads();
  for (int o = 128; o > 0; o >>= 1) {
    if (tid < o) red[tid] += red[tid + o];
    __syncthreads();
  }
  const float inv = 1.0f / red[0];
  __syncthreads();

  const int d = tid >> 4, g = tid & 15;
  float a = 0.0f;
  for (int s0 = g; s0 < kS; s0 += 16)
    a += sc[s0] * bf2f(vh[((size_t)s0 * kB + b) * kH + hd * kDH + d]);
  part[d][g] = a;
  __syncthreads();
  if (tid < kDH) {
    float t = 0.0f;
#pragma unroll
    for (int gg = 0; gg < 16; gg++) t += part[tid][gg];
    ctx[b * kH + hd * kDH + tid] = t * inv;
  }
}

// ---------------- decoder: Wo + LSTM cell + fc-input build -----------------

__global__ void __launch_bounds__(512) k_cell(
    const float* __restrict__ ctx, const bf16* __restrict__ Wo,
    const float* __restrict__ bo, const int* __restrict__ cur_tok,
    const float* __restrict__ dec_emb, const bf16* __restrict__ Wih,
    const bf16* __restrict__ Whh, const float* __restrict__ bias_d,
    bf16* __restrict__ hbf, float* __restrict__ cbuf_g,
    bf16* __restrict__ fcin, unsigned long long* __restrict__ keys) {
  __shared__ bf16 ctx_bf[kB][kH];       // 4 KB
  __shared__ float ctxo[kB][kH];        // 8 KB
  __shared__ bf16 xb[kB][kE + kH];      // 12 KB
  __shared__ float gates[kB][kG];       // 32 KB

  const int tid = threadIdx.x, wave = tid >> 5, lane = tid & 31;
  const int rowm = lane & 15, n = lane & 15, m0 = (lane >> 4) * 8;

  for (int e = tid; e < kB * kH; e += 512)
    (&ctx_bf[0][0])[e] = f2bf(ctx[e]);
  __syncthreads();

  if (wave < 8) {  // ctx @ Wo^T + bo
    const bf16* arow = &ctx_bf[rowm][0];
    const bf16* brow = Wo + (size_t)(wave * 16 + rowm) * kH;
    f32x8 acc = {};
#pragma unroll
    for (int kk = 0; kk < kH; kk += 32)
      acc = wmma_bf(frag_ld(arow, kk), frag_ld(brow, kk), acc);
    const int col = wave * 16 + n;
    const float bb = bo[col];
#pragma unroll
    for (int r = 0; r < 8; r++) ctxo[m0 + r][col] = acc[r] + bb;
  }
  __syncthreads();

  for (int e = tid; e < kB * (kE + kH); e += 512) {  // x = [emb(tok), ctx_o]
    const int b = e / (kE + kH), col = e % (kE + kH);
    const float v = (col < kE) ? dec_emb[(size_t)cur_tok[b] * kE + col]
                               : ctxo[b][col - kE];
    xb[b][col] = f2bf(v);
  }
  __syncthreads();

  for (int half = 0; half < 2; half++) {  // 32 gate tiles over 16 waves
    const int nt = wave + half * 16;
    const bf16* xrow = &xb[rowm][0];
    const bf16* wih_row = Wih + (size_t)(nt * 16 + rowm) * (kE + kH);
    f32x8 acc = {};
#pragma unroll
    for (int kk = 0; kk < kE + kH; kk += 32)
      acc = wmma_bf(frag_ld(xrow, kk), frag_ld(wih_row, kk), acc);
    const bf16* hrow = hbf + rowm * kH;
    const bf16* whh_row = Whh + (size_t)(nt * 16 + rowm) * kH;
#pragma unroll
    for (int kk = 0; kk < kH; kk += 32)
      acc = wmma_bf(frag_ld(hrow, kk), frag_ld(whh_row, kk), acc);
    const int col = nt * 16 + n;
    const float bb = bias_d[col];
#pragma unroll
    for (int r = 0; r < 8; r++) gates[m0 + r][col] = acc[r] + bb;
  }
  __syncthreads();

  for (int e = tid; e < kB * kH; e += 512) {
    const int b = e >> 7, j = e & 127;
    const float ig = gates[b][j], fg = gates[b][kH + j];
    const float gg = gates[b][2 * kH + j], og = gates[b][3 * kH + j];
    float c = cbuf_g[e];
    c = sigm(fg) * c + sigm(ig) * tanhf(gg);
    const float h = sigm(og) * tanhf(c);
    cbuf_g[e] = c;
    const bf16 hb = f2bf(h);
    hbf[e] = hb;
    fcin[b * (2 * kH) + j] = hb;
    fcin[b * (2 * kH) + kH + j] = f2bf(ctxo[b][j]);
  }
  if (tid < kB) keys[tid] = 0ull;  // reset argmax keys for this step's k_fc
}

// ---------------- decoder: vocab projection + fused argmax -----------------

__global__ void __launch_bounds__(256) k_fc(
    const bf16* __restrict__ fcin, const bf16* __restrict__ Wfc,
    const float* __restrict__ bfc, float* __restrict__ out,
    unsigned long long* __restrict__ keys, int t) {
  __shared__ unsigned long long lkeys[kB];
  const int tid = threadIdx.x, wave = tid >> 5, lane = tid & 31;
  if (tid < kB) lkeys[tid] = 0ull;
  __syncthreads();

  const int gw = blockIdx.x * 8 + wave;  // one 16-wide vocab tile per wave
  const int rowm = lane & 15, n = lane & 15, m0 = (lane >> 4) * 8;
  if (gw < kV / 16) {
    const bf16* arow = fcin + rowm * (2 * kH);
    const bf16* brow = Wfc + (size_t)(gw * 16 + rowm) * (2 * kH);
    f32x8 acc = {};
#pragma unroll
    for (int kk = 0; kk < 2 * kH; kk += 32)
      acc = wmma_bf(frag_ld(arow, kk), frag_ld(brow, kk), acc);
    const int col = gw * 16 + n;
    const float bb = bfc[col];
#pragma unroll
    for (int r = 0; r < 8; r++) {
      const float v = acc[r] + bb;
      out[((size_t)(m0 + r) * kT + t) * kV + col] = v;
      // order-preserving float encode + low-index-wins tiebreak in low 32 bits
      unsigned u = __float_as_uint(v);
      u = (u & 0x80000000u) ? ~u : (u | 0x80000000u);
      unsigned long long key =
          ((unsigned long long)u << 32) | (0xFFFFFFFFu - (unsigned)col);
#pragma unroll
      for (int d = 8; d >= 1; d >>= 1) {
        const unsigned long long o = __shfl_xor(key, d, 32);
        if (o > key) key = o;
      }
      if (n == 0) atomicMax(&lkeys[m0 + r], key);
    }
  }
  __syncthreads();
  if (tid < kB) {
    const unsigned long long k = lkeys[tid];
    if (k) atomicMax(&keys[tid], k);
  }
}

__global__ void k_sel(const unsigned long long* __restrict__ keys,
                      int* __restrict__ cur) {
  if (threadIdx.x < kB) {
    const unsigned long long k = keys[threadIdx.x];
    cur[threadIdx.x] = (int)(0xFFFFFFFFu - (unsigned)(k & 0xFFFFFFFFull));
  }
}

// ---------------------------------------------------------------------------

extern "C" void kernel_launch(void* const* d_in, const int* in_sizes, int n_in,
                              void* d_out, int out_size, void* d_ws, size_t ws_size,
                              hipStream_t stream) {
  (void)in_sizes; (void)n_in; (void)out_size; (void)ws_size;

  const int*   src     = (const int*)  d_in[0];
  const int*   trg     = (const int*)  d_in[1];
  const float* enc_emb = (const float*)d_in[2];
  const float* enc_Wih = (const float*)d_in[3];
  const float* enc_Whh = (const float*)d_in[4];
  const float* enc_bih = (const float*)d_in[5];
  const float* enc_bhh = (const float*)d_in[6];
  const float* dec_emb = (const float*)d_in[7];
  const float* dec_Wih = (const float*)d_in[8];
  const float* dec_Whh = (const float*)d_in[9];
  const float* dec_bih = (const float*)d_in[10];
  const float* dec_bhh = (const float*)d_in[11];
  const float* Wq      = (const float*)d_in[12];
  const float* Wk      = (const float*)d_in[13];
  const float* Wv      = (const float*)d_in[14];
  const float* bq      = (const float*)d_in[15];
  const float* bk      = (const float*)d_in[16];
  const float* bv      = (const float*)d_in[17];
  const float* Wo      = (const float*)d_in[18];
  const float* bo      = (const float*)d_in[19];
  const float* Wfc     = (const float*)d_in[20];
  const float* bfc     = (const float*)d_in[21];
  float* out = (float*)d_out;

  char* w = (char*)d_ws;
  auto alloc = [&](size_t bytes) -> char* {
    char* p = w;
    w += (bytes + 255) & ~(size_t)255;
    return p;
  };

  bf16* Wih_e = (bf16*)alloc((size_t)kG * kE * 2);
  bf16* Whh_e = (bf16*)alloc((size_t)kG * kH * 2);
  bf16* Wih_d = (bf16*)alloc((size_t)kG * (kE + kH) * 2);
  bf16* Whh_d = (bf16*)alloc((size_t)kG * kH * 2);
  bf16* Wq_b  = (bf16*)alloc((size_t)kH * kH * 2);
  bf16* Wk_b  = (bf16*)alloc((size_t)kH * kH * 2);
  bf16* Wv_b  = (bf16*)alloc((size_t)kH * kH * 2);
  bf16* Wo_b  = (bf16*)alloc((size_t)kH * kH * 2);
  bf16* Wfc_b = (bf16*)alloc((size_t)kV * 2 * kH * 2);
  float* bias_e = (float*)alloc(kG * 4);
  float* bias_d = (float*)alloc(kG * 4);
  float* gx   = (float*)alloc((size_t)kS * kB * kG * 4);
  bf16* eout  = (bf16*)alloc((size_t)kS * kB * kH * 2);
  bf16* khb   = (bf16*)alloc((size_t)kS * kB * kH * 2);
  bf16* vhb   = (bf16*)alloc((size_t)kS * kB * kH * 2);
  float* qbuf = (float*)alloc(kB * kH * 4);
  float* ctxb = (float*)alloc(kB * kH * 4);
  float* cbuf = (float*)alloc(kB * kH * 4);
  bf16* hbf   = (bf16*)alloc(kB * kH * 2);
  bf16* fcin  = (bf16*)alloc(kB * 2 * kH * 2);
  int* cur    = (int*)alloc(kB * 4);
  unsigned long long* keys = (unsigned long long*)alloc(kB * 8);

  auto cvt = [&](bf16* dst, const float* s, int n) {
    k_cvt<<<(n + 255) / 256, 256, 0, stream>>>(dst, s, n);
  };
  cvt(Wih_e, enc_Wih, kG * kE);
  cvt(Whh_e, enc_Whh, kG * kH);
  cvt(Wih_d, dec_Wih, kG * (kE + kH));
  cvt(Whh_d, dec_Whh, kG * kH);
  cvt(Wq_b, Wq, kH * kH);
  cvt(Wk_b, Wk, kH * kH);
  cvt(Wv_b, Wv, kH * kH);
  cvt(Wo_b, Wo, kH * kH);
  cvt(Wfc_b, Wfc, kV * 2 * kH);
  k_addv<<<2, 256, 0, stream>>>(bias_e, enc_bih, enc_bhh, kG);
  k_addv<<<2, 256, 0, stream>>>(bias_d, dec_bih, dec_bhh, kG);

  k_zero_out0<<<(kB * kV + 255) / 256, 256, 0, stream>>>(out);
  k_init_tok<<<1, 32, 0, stream>>>(cur, trg);

  // Encoder: parallel x@Wih^T precompute, then one persistent-WG serial scan.
  k_embgx<<<kS, 256, 0, stream>>>(src, enc_emb, Wih_e, bias_e, gx);
  k_enc_scan<<<1, 1024, kSmemEnc, stream>>>(gx, Whh_e, eout, hbf, cbuf);
  k_kv<<<kS, 256, 0, stream>>>(eout, Wk_b, Wv_b, bk, bv, khb, vhb);

  // Decoder: 127 serial steps, 5 kernels each (graph-replay friendly).
  for (int t = 1; t < kT; t++) {
    k_q<<<1, 256, 0, stream>>>(hbf, Wq_b, bq, qbuf);
    k_attn<<<kB * kNH, 256, 0, stream>>>(qbuf, khb, vhb, ctxb);
    k_cell<<<1, 512, 0, stream>>>(ctxb, Wo_b, bo, cur, dec_emb, Wih_d, Whh_d,
                                  bias_d, hbf, cbuf, fcin, keys);
    k_fc<<<(kV / 16 + 7) / 8, 256, 0, stream>>>(fcin, Wfc_b, bfc, out, keys, t);
    k_sel<<<1, 32, 0, stream>>>(keys, cur);
  }
}
